// TopkCompetitiveInstanceAttention_6768868458557
// MI455X (gfx1250) — matrix operand order repaired and checked by reading, not compiled, not run on hardware
//
#include <hip/hip_runtime.h>

typedef __bf16 bf16;
typedef __attribute__((ext_vector_type(16))) __bf16 v16bf;
typedef __attribute__((ext_vector_type(8)))  __bf16 v8bf;
typedef __attribute__((ext_vector_type(8)))  float  v8f;

#define NB 16
#define NQ 128
#define NK 4096
#define ND 512
#define NH 8
#define NC 64
#define NKEEP 33
#define NKPAD 48

__device__ __forceinline__ v8f v8f_zero() {
  v8f z;
#pragma unroll
  for (int i = 0; i < 8; ++i) z[i] = 0.0f;
  return z;
}

__device__ __forceinline__ v8f wmma_bf16(v16bf a, v16bf b, v8f c) {
  return __builtin_amdgcn_wmma_f32_16x16x32_bf16(false, a, false, b, (short)0, c,
                                                 false, false);
}

// A fragment, 16x32 bf16, source row-major [m][k], ld in elements.
// lane&15 = M row; lane>=16 owns k-offsets +16. 16 contiguous bf16 per lane.
__device__ __forceinline__ v16bf load_a_frag(const bf16* base, int ld, int m0, int k0) {
  const int lane = threadIdx.x & 31;
  const bf16* p = base + (m0 + (lane & 15)) * ld + k0 + (((lane >> 4)) << 4);
  v8bf lo = *(const v8bf*)(p);
  v8bf hi = *(const v8bf*)(p + 8);
  return __builtin_shufflevector(lo, hi, 0, 1, 2, 3, 4, 5, 6, 7, 8, 9, 10, 11, 12, 13, 14, 15);
}

// B fragment, logical B[k][n] stored as src[n][k] row-major ("transposed" source).
// lane&15 = N column; lane>=16 owns k-offsets +16. 16 contiguous bf16 per lane.
__device__ __forceinline__ v16bf load_bt_frag(const bf16* base, int ld, int n0, int k0) {
  const int lane = threadIdx.x & 31;
  const bf16* p = base + (n0 + (lane & 15)) * ld + k0 + ((lane >> 4) << 4);
  v8bf lo = *(const v8bf*)(p);
  v8bf hi = *(const v8bf*)(p + 8);
  return __builtin_shufflevector(lo, hi, 0, 1, 2, 3, 4, 5, 6, 7, 8, 9, 10, 11, 12, 13, 14, 15);
}

// ---------------------------------------------------------------- utilities
__global__ void cvt_bf16_kernel(const float* __restrict__ src, bf16* __restrict__ dst, int n) {
  int i = blockIdx.x * 256 + threadIdx.x;
  if (i < n) dst[i] = (bf16)src[i];
}

__global__ void zero_i32_kernel(int* __restrict__ p, int n) {
  int i = blockIdx.x * 256 + threadIdx.x;
  if (i < n) p[i] = 0;
}

// ------------------------------------------------- q projection (scaled by 1/8)
// out[m][n] = (1/8) * sum_k (A0+A1)[m][k] * W[n][k];  M=2048, N=K=512
__global__ void proj_q_kernel(const float* __restrict__ A0, const float* __restrict__ A1,
                              const bf16* __restrict__ Wbf, bf16* __restrict__ out) {
  __shared__ bf16 As[64 * 32];
  __shared__ bf16 Bs[64 * 32];
  const int m0 = blockIdx.x * 64, n0 = blockIdx.y * 64;
  const int wave = threadIdx.x >> 5, lane = threadIdx.x & 31;
  const int wr = wave >> 1, wc = wave & 1;
  v8f acc0 = v8f_zero(), acc1 = v8f_zero();
  for (int k0 = 0; k0 < ND; k0 += 32) {
    for (int t = threadIdx.x; t < 64 * 32; t += 256) {
      int r = t >> 5, c = t & 31;
      int gi = (m0 + r) * ND + k0 + c;
      As[t] = (bf16)(A0[gi] + A1[gi]);
      Bs[t] = Wbf[(n0 + r) * ND + k0 + c];
    }
    __syncthreads();
    v16bf a  = load_a_frag(As, 32, wr * 16, 0);
    v16bf b0 = load_bt_frag(Bs, 32, wc * 32, 0);
    v16bf b1 = load_bt_frag(Bs, 32, wc * 32 + 16, 0);
    acc0 = wmma_bf16(a, b0, acc0);
    acc1 = wmma_bf16(a, b1, acc1);
    __syncthreads();
  }
  const int rbase = m0 + wr * 16 + ((lane >> 4) << 3);
  const int cbase = n0 + wc * 32 + (lane & 15);
#pragma unroll
  for (int v = 0; v < 8; ++v) {
    out[(rbase + v) * ND + cbase]      = (bf16)(acc0[v] * 0.125f);
    out[(rbase + v) * ND + cbase + 16] = (bf16)(acc1[v] * 0.125f);
  }
}

// --------------------------------------- fused k / vf / ve projections (M=65536)
__global__ void proj_kve_kernel(const float* __restrict__ ctx, const float* __restrict__ enc,
                                const bf16* __restrict__ Wk, const bf16* __restrict__ Wf,
                                const bf16* __restrict__ We, bf16* __restrict__ kproj,
                                bf16* __restrict__ vfp, bf16* __restrict__ vep) {
  __shared__ bf16 Cs[64 * 32], Es[64 * 32], Ss[64 * 32];
  __shared__ bf16 Bk[64 * 32], Bf[64 * 32], Be[64 * 32];
  const int m0 = blockIdx.x * 64, n0 = blockIdx.y * 64;
  const int wave = threadIdx.x >> 5, lane = threadIdx.x & 31;
  const int wr = wave >> 1, wc = wave & 1;
  v8f aK0 = v8f_zero(), aK1 = v8f_zero();
  v8f aF0 = v8f_zero(), aF1 = v8f_zero();
  v8f aE0 = v8f_zero(), aE1 = v8f_zero();
  for (int k0 = 0; k0 < ND; k0 += 32) {
    for (int t = threadIdx.x; t < 64 * 32; t += 256) {
      int r = t >> 5, c = t & 31;
      int gi = (m0 + r) * ND + k0 + c;
      float cv = ctx[gi], ev = enc[gi];
      Cs[t] = (bf16)cv;
      Es[t] = (bf16)ev;
      Ss[t] = (bf16)(cv + ev);
      int wi = (n0 + r) * ND + k0 + c;
      Bk[t] = Wk[wi];
      Bf[t] = Wf[wi];
      Be[t] = We[wi];
    }
    __syncthreads();
    v16bf as = load_a_frag(Ss, 32, wr * 16, 0);
    v16bf ac = load_a_frag(Cs, 32, wr * 16, 0);
    v16bf ae = load_a_frag(Es, 32, wr * 16, 0);
    v16bf bk0 = load_bt_frag(Bk, 32, wc * 32, 0);
    v16bf bk1 = load_bt_frag(Bk, 32, wc * 32 + 16, 0);
    v16bf bf0 = load_bt_frag(Bf, 32, wc * 32, 0);
    v16bf bf1 = load_bt_frag(Bf, 32, wc * 32 + 16, 0);
    v16bf be0 = load_bt_frag(Be, 32, wc * 32, 0);
    v16bf be1 = load_bt_frag(Be, 32, wc * 32 + 16, 0);
    aK0 = wmma_bf16(as, bk0, aK0); aK1 = wmma_bf16(as, bk1, aK1);
    aF0 = wmma_bf16(ac, bf0, aF0); aF1 = wmma_bf16(ac, bf1, aF1);
    aE0 = wmma_bf16(ae, be0, aE0); aE1 = wmma_bf16(ae, be1, aE1);
    __syncthreads();
  }
  const int rbase = m0 + wr * 16 + ((lane >> 4) << 3);
  const int cbase = n0 + wc * 32 + (lane & 15);
#pragma unroll
  for (int v = 0; v < 8; ++v) {
    int r0 = (rbase + v) * ND + cbase;
    kproj[r0]      = (bf16)aK0[v]; kproj[r0 + 16] = (bf16)aK1[v];
    vfp[r0]        = (bf16)aF0[v]; vfp[r0 + 16]   = (bf16)aF1[v];
    vep[r0]        = (bf16)aE0[v]; vep[r0 + 16]   = (bf16)aE1[v];
  }
}

// ----------------------------------------- pass 1: query-axis softmax scoring
// score[b][q-1] = max over (h,k) of softmax_{q'=1..127}(dots)[q]  (int-bit atomics)
__global__ void score_kernel(const bf16* __restrict__ qproj, const bf16* __restrict__ kproj,
                             int* __restrict__ score) {
  const int b = blockIdx.x >> 3, h = blockIdx.x & 7;
  __shared__ bf16 Qs[NQ * NC];   // 16 KB
  __shared__ bf16 Ks[NQ * NC];   // 16 KB (128-key tile)
  __shared__ int scoreLDS[NQ];
  const int wave = threadIdx.x >> 5, lane = threadIdx.x & 31;

  for (int t = threadIdx.x; t < NQ * NC; t += 256) {
    int q = t >> 6, c = t & 63;
    Qs[t] = qproj[(b * NQ + q) * ND + h * NC + c];
  }
  if (threadIdx.x < NQ) scoreLDS[threadIdx.x] = 0;
  __syncthreads();

  const int rowoff = (lane >> 4) << 3;
  float rm[8][8];
#pragma unroll
  for (int mt = 0; mt < 8; ++mt)
#pragma unroll
    for (int v = 0; v < 8; ++v) rm[mt][v] = 0.0f;

  for (int kt = 0; kt < NK; kt += 128) {
    for (int t = threadIdx.x; t < 128 * NC; t += 256) {
      int kk = t >> 6, c = t & 63;
      Ks[t] = kproj[((b * NK) + kt + kk) * ND + h * NC + c];
    }
    __syncthreads();
    v8f acc[8];
#pragma unroll
    for (int mt = 0; mt < 8; ++mt) acc[mt] = v8f_zero();
#pragma unroll
    for (int c0 = 0; c0 < NC; c0 += 32) {
      v16bf bfr = load_bt_frag(Ks, NC, wave * 16, c0);
#pragma unroll
      for (int mt = 0; mt < 8; ++mt) {
        v16bf a = load_a_frag(Qs, NC, mt * 16, c0);
        acc[mt] = wmma_bf16(a, bfr, acc[mt]);
      }
    }
    // per-key softmax over queries 1..127 (column lives in lanes n and n+16)
    float cmax = -3.0e38f;
#pragma unroll
    for (int mt = 0; mt < 8; ++mt)
#pragma unroll
      for (int v = 0; v < 8; ++v) {
        int row = mt * 16 + v + rowoff;
        if (row != 0) cmax = fmaxf(cmax, acc[mt][v]);
      }
    cmax = fmaxf(cmax, __shfl_xor(cmax, 16, 32));
    float csum = 0.0f;
#pragma unroll
    for (int mt = 0; mt < 8; ++mt)
#pragma unroll
      for (int v = 0; v < 8; ++v) {
        int row = mt * 16 + v + rowoff;
        if (row != 0) csum += __expf(acc[mt][v] - cmax);
      }
    csum += __shfl_xor(csum, 16, 32);
    float inv = 1.0f / csum;
#pragma unroll
    for (int mt = 0; mt < 8; ++mt)
#pragma unroll
      for (int v = 0; v < 8; ++v) {
        int row = mt * 16 + v + rowoff;
        if (row != 0) {
          float p = __expf(acc[mt][v] - cmax) * inv;
          rm[mt][v] = fmaxf(rm[mt][v], p);
        }
      }
    __syncthreads();
  }
  // reduce running max over the 16 key-columns held per half-wave
#pragma unroll
  for (int d = 1; d < 16; d <<= 1)
#pragma unroll
    for (int mt = 0; mt < 8; ++mt)
#pragma unroll
      for (int v = 0; v < 8; ++v) rm[mt][v] = fmaxf(rm[mt][v], __shfl_xor(rm[mt][v], d, 32));
  if ((lane & 15) == 0) {
#pragma unroll
    for (int mt = 0; mt < 8; ++mt)
#pragma unroll
      for (int v = 0; v < 8; ++v) {
        int row = mt * 16 + v + rowoff;
        if (row > 0) atomicMax(&scoreLDS[row], __float_as_int(rm[mt][v]));
      }
  }
  __syncthreads();
  if (threadIdx.x > 0 && threadIdx.x < NQ)
    atomicMax(&score[b * (NQ - 1) + threadIdx.x - 1], scoreLDS[threadIdx.x]);
}

// ----------------------------- top-k (MQ=32, keep_first) + gather kept queries
__global__ void topk_gather_kernel(const int* __restrict__ score, const bf16* __restrict__ qproj,
                                   float* __restrict__ keep_out, bf16* __restrict__ qkept) {
  const int b = blockIdx.x;
  __shared__ float s[NQ - 1];
  __shared__ int kidx[NKPAD];
  if (threadIdx.x < NQ - 1) s[threadIdx.x] = __int_as_float(score[b * (NQ - 1) + threadIdx.x]);
  __syncthreads();
  if (threadIdx.x == 0) {
    kidx[0] = 0;
    for (int i = 0; i < 32; ++i) {
      float best = -1.0f; int bj = 0;
      for (int j = 0; j < NQ - 1; ++j)
        if (s[j] > best) { best = s[j]; bj = j; }   // strict '>' => lowest-index tiebreak
      kidx[i + 1] = bj + 1;
      s[bj] = -2.0f;
    }
    for (int i = NKEEP; i < NKPAD; ++i) kidx[i] = 0;
  }
  __syncthreads();
  if (threadIdx.x < NKEEP) keep_out[b * NKEEP + threadIdx.x] = (float)kidx[threadIdx.x];
  for (int t = threadIdx.x; t < NH * NKPAD * NC; t += 256) {
    int h = t / (NKPAD * NC);
    int r = (t / NC) % NKPAD;
    int c = t & (NC - 1);
    bf16 val = (r < NKEEP) ? qproj[(b * NQ + kidx[r]) * ND + h * NC + c] : (bf16)0.0f;
    qkept[((b * NH + h) * NKPAD + r) * NC + c] = val;
  }
}

// ------------------ pass 2: competitive softmax attention for 33 kept queries
__global__ void attn_kernel(const bf16* __restrict__ qkept, const bf16* __restrict__ kproj,
                            const bf16* __restrict__ vfp, const bf16* __restrict__ vep,
                            bf16* __restrict__ outf, bf16* __restrict__ oute) {
  const int b = blockIdx.x >> 3, h = blockIdx.x & 7;
  __shared__ bf16 Qs[NKPAD * NC];   //  6 KB
  __shared__ bf16 Xs[128 * NC];     // 16 KB: k-tile, then reused for P[48][128]
  __shared__ bf16 VfT[NC * 128];    // 16 KB (transposed)
  __shared__ bf16 VeT[NC * 128];    // 16 KB (transposed)
  __shared__ float denom[NKPAD];
  const int wave = threadIdx.x >> 5, lane = threadIdx.x & 31;
  const int isE = wave >> 2;
  const int c0 = (wave & 3) * 16;
  const int rowoff = (lane >> 4) << 3;

  for (int t = threadIdx.x; t < NKPAD * NC; t += 256)
    Qs[t] = qkept[(b * NH + h) * (NKPAD * NC) + t];
  if (threadIdx.x < NKPAD) denom[threadIdx.x] = 0.0f;
  v8f pv0 = v8f_zero(), pv1 = v8f_zero(), pv2 = v8f_zero();
  __syncthreads();

  for (int kt = 0; kt < NK; kt += 128) {
    for (int t = threadIdx.x; t < 128 * NC; t += 256) {
      int kk = t >> 6, c = t & 63;
      int gi = ((b * NK) + kt + kk) * ND + h * NC + c;
      Xs[t] = kproj[gi];
      VfT[c * 128 + kk] = vfp[gi];
      VeT[c * 128 + kk] = vep[gi];
    }
    __syncthreads();

    v8f s0 = v8f_zero(), s1 = v8f_zero(), s2 = v8f_zero();
#pragma unroll
    for (int cc = 0; cc < NC; cc += 32) {
      v16bf bfr = load_bt_frag(Xs, NC, wave * 16, cc);
      s0 = wmma_bf16(load_a_frag(Qs, NC, 0, cc), bfr, s0);
      s1 = wmma_bf16(load_a_frag(Qs, NC, 16, cc), bfr, s1);
      s2 = wmma_bf16(load_a_frag(Qs, NC, 32, cc), bfr, s2);
    }
    float sv[3][8];
#pragma unroll
    for (int v = 0; v < 8; ++v) { sv[0][v] = s0[v]; sv[1][v] = s1[v]; sv[2][v] = s2[v]; }

    // per-key softmax over the 33 kept queries (rows 33..47 masked out)
    float cmax = -3.0e38f;
#pragma unroll
    for (int mt = 0; mt < 3; ++mt)
#pragma unroll
      for (int v = 0; v < 8; ++v) {
        int row = mt * 16 + v + rowoff;
        if (row < NKEEP) cmax = fmaxf(cmax, sv[mt][v]);
      }
    cmax = fmaxf(cmax, __shfl_xor(cmax, 16, 32));
    float csum = 0.0f;
#pragma unroll
    for (int mt = 0; mt < 3; ++mt)
#pragma unroll
      for (int v = 0; v < 8; ++v) {
        int row = mt * 16 + v + rowoff;
        if (row < NKEEP) csum += __expf(sv[mt][v] - cmax);
      }
    csum += __shfl_xor(csum, 16, 32);
    float inv = 1.0f / csum;
    float p[3][8];
#pragma unroll
    for (int mt = 0; mt < 3; ++mt)
#pragma unroll
      for (int v = 0; v < 8; ++v) {
        int row = mt * 16 + v + rowoff;
        p[mt][v] = (row < NKEEP) ? __expf(sv[mt][v] - cmax) * inv : 0.0f;
      }
    __syncthreads();   // all waves finished reading Xs as the k-tile

    // write P (bf16) into Xs as P[48][128]; this lane owns one key column
    const int ncol = wave * 16 + (lane & 15);
#pragma unroll
    for (int mt = 0; mt < 3; ++mt)
#pragma unroll
      for (int v = 0; v < 8; ++v) {
        int row = mt * 16 + v + rowoff;
        Xs[row * 128 + ncol] = (bf16)p[mt][v];
      }
    // denominator partials: sum p over this wave's 16 key columns
    float q[3][8];
#pragma unroll
    for (int mt = 0; mt < 3; ++mt)
#pragma unroll
      for (int v = 0; v < 8; ++v) q[mt][v] = p[mt][v];
#pragma unroll
    for (int d = 1; d < 16; d <<= 1)
#pragma unroll
      for (int mt = 0; mt < 3; ++mt)
#pragma unroll
        for (int v = 0; v < 8; ++v) q[mt][v] += __shfl_xor(q[mt][v], d, 32);
    if ((lane & 15) == 0) {
#pragma unroll
      for (int mt = 0; mt < 3; ++mt)
#pragma unroll
        for (int v = 0; v < 8; ++v) {
          int row = mt * 16 + v + rowoff;
          if (row < NKEEP) atomicAdd(&denom[row], q[mt][v]);
        }
    }
    __syncthreads();   // P complete

    // P @ V : waves 0-3 -> vf c-strips, waves 4-7 -> ve c-strips
    const bf16* VT = isE ? VeT : VfT;
#pragma unroll
    for (int kk2 = 0; kk2 < 128; kk2 += 32) {
      v16bf bv = load_bt_frag(VT, 128, c0, kk2);
      pv0 = wmma_bf16(load_a_frag(Xs, 128, 0, kk2), bv, pv0);
      pv1 = wmma_bf16(load_a_frag(Xs, 128, 16, kk2), bv, pv1);
      pv2 = wmma_bf16(load_a_frag(Xs, 128, 32, kk2), bv, pv2);
    }
    __syncthreads();   // before next tile overwrites staging
  }

  bf16* outp = isE ? oute : outf;
  const int col = h * NC + c0 + (lane & 15);
  float pvv[3][8];
#pragma unroll
  for (int v = 0; v < 8; ++v) { pvv[0][v] = pv0[v]; pvv[1][v] = pv1[v]; pvv[2][v] = pv2[v]; }
#pragma unroll
  for (int mt = 0; mt < 3; ++mt)
#pragma unroll
    for (int v = 0; v < 8; ++v) {
      int row = mt * 16 + v + rowoff;
      if (row < NKEEP) {
        float dd = denom[row] + 1e-8f;
        outp[(b * NKEEP + row) * ND + col] = (bf16)(pvv[mt][v] / dd);
      }
    }
}

// ----------------------------------- final output projections (+bias, fp32 out)
__global__ void outproj_kernel(const bf16* __restrict__ attf, const bf16* __restrict__ atte,
                               const bf16* __restrict__ Wof, const bf16* __restrict__ Woe,
                               const float* __restrict__ bof, const float* __restrict__ boe,
                               float* __restrict__ dout) {
  const int m0 = blockIdx.x * 16, n0 = blockIdx.y * 16;
  const int e = blockIdx.z;
  const bf16* A = e ? atte : attf;
  const bf16* W = e ? Woe : Wof;
  const float* bias = e ? boe : bof;
  float* out = dout + (size_t)e * (NB * NKEEP * ND);
  v8f acc = v8f_zero();
  for (int k0 = 0; k0 < ND; k0 += 32) {
    v16bf a = load_a_frag(A, ND, m0, k0);
    v16bf bw = load_bt_frag(W, ND, n0, k0);
    acc = wmma_bf16(a, bw, acc);
  }
  const int lane = threadIdx.x & 31;
  const int col = n0 + (lane & 15);
  const int rbase = m0 + ((lane >> 4) << 3);
  const float bv = bias[col];
#pragma unroll
  for (int v = 0; v < 8; ++v) out[(rbase + v) * ND + col] = acc[v] + bv;
}

// ---------------------------------------------------------------------------
extern "C" void kernel_launch(void* const* d_in, const int* in_sizes, int n_in,
                              void* d_out, int out_size, void* d_ws, size_t ws_size,
                              hipStream_t stream) {
  const float* slots     = (const float*)d_in[0];
  const float* slots_enc = (const float*)d_in[1];
  const float* ctx       = (const float*)d_in[2];
  const float* ctx_enc   = (const float*)d_in[3];
  const float* Wq  = (const float*)d_in[4];
  const float* Wk  = (const float*)d_in[5];
  const float* Wf  = (const float*)d_in[6];
  const float* We  = (const float*)d_in[7];
  const float* Wof = (const float*)d_in[8];
  const float* bof = (const float*)d_in[9];
  const float* Woe = (const float*)d_in[10];
  const float* boe = (const float*)d_in[11];
  float* out = (float*)d_out;

  size_t off = 0;
  char* base = (char*)d_ws;
  auto wsalloc = [&](size_t bytes) -> void* {
    void* p = (void*)(base + off);
    off += (bytes + 255) & ~(size_t)255;
    return p;
  };
  const size_t WSZ = (size_t)ND * ND * sizeof(bf16);
  bf16* wq_bf  = (bf16*)wsalloc(WSZ);
  bf16* wk_bf  = (bf16*)wsalloc(WSZ);
  bf16* wf_bf  = (bf16*)wsalloc(WSZ);
  bf16* we_bf  = (bf16*)wsalloc(WSZ);
  bf16* wof_bf = (bf16*)wsalloc(WSZ);
  bf16* woe_bf = (bf16*)wsalloc(WSZ);
  bf16* qproj  = (bf16*)wsalloc((size_t)NB * NQ * ND * sizeof(bf16));
  bf16* kproj  = (bf16*)wsalloc((size_t)NB * NK * ND * sizeof(bf16));
  bf16* vfp    = (bf16*)wsalloc((size_t)NB * NK * ND * sizeof(bf16));
  bf16* vep    = (bf16*)wsalloc((size_t)NB * NK * ND * sizeof(bf16));
  int*  score  = (int*)wsalloc((size_t)NB * (NQ - 1) * sizeof(int));
  bf16* qkept  = (bf16*)wsalloc((size_t)NB * NH * NKPAD * NC * sizeof(bf16));
  bf16* attf   = (bf16*)wsalloc((size_t)NB * NKEEP * ND * sizeof(bf16));
  bf16* atte   = (bf16*)wsalloc((size_t)NB * NKEEP * ND * sizeof(bf16));

  const int WN = ND * ND;
  cvt_bf16_kernel<<<(WN + 255) / 256, 256, 0, stream>>>(Wq, wq_bf, WN);
  cvt_bf16_kernel<<<(WN + 255) / 256, 256, 0, stream>>>(Wk, wk_bf, WN);
  cvt_bf16_kernel<<<(WN + 255) / 256, 256, 0, stream>>>(Wf, wf_bf, WN);
  cvt_bf16_kernel<<<(WN + 255) / 256, 256, 0, stream>>>(We, we_bf, WN);
  cvt_bf16_kernel<<<(WN + 255) / 256, 256, 0, stream>>>(Wof, wof_bf, WN);
  cvt_bf16_kernel<<<(WN + 255) / 256, 256, 0, stream>>>(Woe, woe_bf, WN);
  zero_i32_kernel<<<(NB * (NQ - 1) + 255) / 256, 256, 0, stream>>>(score, NB * (NQ - 1));

  proj_q_kernel<<<dim3((NB * NQ) / 64, ND / 64), 256, 0, stream>>>(slots, slots_enc, wq_bf, qproj);
  proj_kve_kernel<<<dim3((NB * NK) / 64, ND / 64), 256, 0, stream>>>(
      ctx, ctx_enc, wk_bf, wf_bf, we_bf, kproj, vfp, vep);
  score_kernel<<<NB * NH, 256, 0, stream>>>(qproj, kproj, score);

  float* keep_out = out + (size_t)2 * NB * NKEEP * ND;
  topk_gather_kernel<<<NB, 256, 0, stream>>>(score, qproj, keep_out, qkept);
  attn_kernel<<<NB * NH, 256, 0, stream>>>(qkept, kproj, vfp, vep, attf, atte);
  outproj_kernel<<<dim3((NB * NKEEP) / 33 * 33 / 16, ND / 16, 2), 32, 0, stream>>>(
      attf, atte, wof_bf, woe_bf, bof, boe, out);
}
